// CausalSelfAttention_22763326669141
// MI455X (gfx1250) — compile-verified, matrix operand order checked by reading, not run on hardware
//
#include <hip/hip_runtime.h>
#include <hip/hip_bf16.h>
#include <stdint.h>

// ---------------------------------------------------------------------------
// CausalSelfAttention forward for MI455X (gfx1250, wave32, WMMA).
//   B=4, T=2048, C=1024, H=16, D=64.  All matmuls via v_wmma_f32_16x16x32_f16.
// Pipeline: cvt(x,W) -> GEMM(qkv) -> flash-attn -> GEMM(proj, f32 out)
// Data staging uses GLOBAL_LOAD_ASYNC_TO_LDS (ASYNCcnt) when the toolchain
// exposes it, with a synchronous fallback otherwise.
// ---------------------------------------------------------------------------

typedef _Float16 h16;
typedef __attribute__((ext_vector_type(16))) _Float16 v16h;
typedef __attribute__((ext_vector_type(8)))  _Float16 v8h;
typedef __attribute__((ext_vector_type(8)))  float    v8f;
typedef int v4i __attribute__((vector_size(4 * sizeof(int))));

#define BATCH   4
#define SEQ     2048
#define N_EMBD  1024
#define N_HEAD  16
#define HEAD_D  64

#if defined(__HIP_DEVICE_COMPILE__) && \
    __has_builtin(__builtin_amdgcn_global_load_async_to_lds_b128)
#define ASYNC_LDS 1
#else
#define ASYNC_LDS 0
#endif

// 16-byte global -> LDS stage.  Async path goes straight to LDS (no VGPR
// round-trip) and is tracked by ASYNCcnt; sync fallback is a plain copy.
// Builtin signature (from compiler diagnostic): arg0 = v4i in addrspace(1),
// arg1 = v4i in addrspace(3), then i32 offset, i32 cpol.
__device__ __forceinline__ void stage16(h16* lds_dst, const h16* gsrc) {
#if ASYNC_LDS
  // Flat LDS address: low 32 bits are the LDS offset.  Flat global address
  // equals the global address.  Integer round-trips always compile.
  __builtin_amdgcn_global_load_async_to_lds_b128(
      (__attribute__((address_space(1))) v4i*)(uintptr_t)gsrc,
      (__attribute__((address_space(3))) v4i*)(uint32_t)(uintptr_t)lds_dst,
      0, 0);
#else
  *(v8h*)lds_dst = *(const v8h*)gsrc;
#endif
}

__device__ __forceinline__ void wait_async_lds() {
#if ASYNC_LDS
#if __has_builtin(__builtin_amdgcn_s_wait_asynccnt)
  __builtin_amdgcn_s_wait_asynccnt(0);
#else
  asm volatile("s_wait_asynccnt 0x0" ::: "memory");
#endif
#endif
}

// Load one 16x32 f16 WMMA operand fragment from a row-major [rows][ld] buffer.
// base points at (row 0, k 0) of the tile.  CDNA5 16-bit A/B lane layout:
//   lanes 0-15 : M/N = lane,    K in {0..7} U {16..23}
//   lanes 16-31: M/N = lane-16, K in {8..15} U {24..31}
// -> two contiguous 16-byte chunks per lane.
__device__ __forceinline__ v16h load_frag(const h16* base, int ld) {
  const int l = threadIdx.x & 31;
  const h16* p = base + (l & 15) * ld + ((l >> 4) << 3);
  v8h lo = *(const v8h*)(p);
  v8h hi = *(const v8h*)(p + 16);
  v16h out;
#pragma unroll
  for (int i = 0; i < 8; ++i) { out[i] = lo[i]; out[i + 8] = hi[i]; }
  return out;
}

// ---------------------------------------------------------------------------
// fp32 -> fp16 convert (grid-stride)
// ---------------------------------------------------------------------------
__global__ void cvt_f32_f16(const float* __restrict__ in, h16* __restrict__ out,
                            size_t n) {
  size_t i = (size_t)blockIdx.x * blockDim.x + threadIdx.x;
  size_t stride = (size_t)gridDim.x * blockDim.x;
  for (; i < n; i += stride) out[i] = (h16)in[i];
}

// ---------------------------------------------------------------------------
// fp32 [K][N] -> fp16 [N][K] transposed convert (32x32 LDS tiles)
// ---------------------------------------------------------------------------
__global__ void transpose_cvt(const float* __restrict__ W, h16* __restrict__ Wt,
                              int K, int N) {
  __shared__ float tile[32][33];
  const int n0 = blockIdx.x * 32, k0 = blockIdx.y * 32;
  const int tx = threadIdx.x, ty = threadIdx.y;  // 32 x 8
#pragma unroll
  for (int r = 0; r < 32; r += 8)
    tile[ty + r][tx] = W[(size_t)(k0 + ty + r) * N + n0 + tx];
  __syncthreads();
#pragma unroll
  for (int r = 0; r < 32; r += 8)
    Wt[(size_t)(n0 + ty + r) * K + k0 + tx] = (h16)tile[tx][ty + r];
}

// ---------------------------------------------------------------------------
// C[M][N] (f32 acc) = A[M][K] (f16, row-major) x Bt[N][K] (f16, B transposed)
// Block tile 128x128, 8 waves of 64x32 (4x2 frags each).  K staged 32-deep,
// double-buffered: tile k+1 is async-copied to LDS while WMMA consumes tile k.
// ---------------------------------------------------------------------------
template <typename OutT>
__global__ __launch_bounds__(256)
void gemm_f16(const h16* __restrict__ A, const h16* __restrict__ Bt,
              OutT* __restrict__ C, int M, int N, int K) {
  __shared__ alignas(16) h16 As[2][128 * 32];
  __shared__ alignas(16) h16 Bs[2][128 * 32];
  const int tid  = threadIdx.x;
  const int wid  = tid >> 5;
  const int lane = tid & 31;
  const int m0 = blockIdx.y * 128;
  const int n0 = blockIdx.x * 128;
  const int wm = (wid >> 2) * 64;  // 2 wave rows
  const int wn = (wid & 3) * 32;   // 4 wave cols

  auto stage_tile = [&](int buf, int k0) {
#pragma unroll
    for (int c = 0; c < 2; ++c) {
      int idx = tid + c * 256;  // 0..511 chunks of 8 f16
      int row = idx >> 2;
      int ko  = (idx & 3) * 8;
      stage16(As[buf] + row * 32 + ko, A + (size_t)(m0 + row) * K + k0 + ko);
      stage16(Bs[buf] + row * 32 + ko, Bt + (size_t)(n0 + row) * K + k0 + ko);
    }
  };

  v8f acc[4][2];
#pragma unroll
  for (int i = 0; i < 4; ++i)
#pragma unroll
    for (int j = 0; j < 2; ++j) acc[i][j] = (v8f){0, 0, 0, 0, 0, 0, 0, 0};

  stage_tile(0, 0);
  int buf = 0;
  for (int k0 = 0; k0 < K; k0 += 32) {
    wait_async_lds();    // this wave's copies into `buf` are in LDS
    __syncthreads();     // ... and everyone else's too
    if (k0 + 32 < K) stage_tile(buf ^ 1, k0 + 32);  // overlap next-tile copy

    v16h af[4], bf[2];
#pragma unroll
    for (int i = 0; i < 4; ++i)
      af[i] = load_frag(As[buf] + (wm + i * 16) * 32, 32);
#pragma unroll
    for (int j = 0; j < 2; ++j)
      bf[j] = load_frag(Bs[buf] + (wn + j * 16) * 32, 32);
#pragma unroll
    for (int i = 0; i < 4; ++i)
#pragma unroll
      for (int j = 0; j < 2; ++j)
        acc[i][j] = __builtin_amdgcn_wmma_f32_16x16x32_f16(
            false, af[i], false, bf[j], (short)0, acc[i][j], false, false);
    buf ^= 1;
  }

  // C/D layout: vgpr r -> row r (lanes 0-15) / row 8+r (lanes 16-31), col=lane&15
  const int rbase = (lane >> 4) << 3;
  const int cin   = lane & 15;
#pragma unroll
  for (int i = 0; i < 4; ++i)
#pragma unroll
    for (int j = 0; j < 2; ++j)
#pragma unroll
      for (int r = 0; r < 8; ++r) {
        int row = m0 + wm + i * 16 + rbase + r;
        int col = n0 + wn + j * 16 + cin;
        C[(size_t)row * N + col] = (OutT)acc[i][j][r];
      }
}

// ---------------------------------------------------------------------------
// Flash attention: one block per (b, h, 64-query tile); 4 waves x 16 q-rows.
// qkv layout: [B][T][3C], q at col h*64, k at 1024+h*64, v at 2048+h*64.
// ---------------------------------------------------------------------------
__global__ __launch_bounds__(128)
void flash_attn(const h16* __restrict__ qkv, const int* __restrict__ amask,
                h16* __restrict__ y) {
  const int qt   = blockIdx.x;  // 64-row query tile index
  const int h    = blockIdx.y;
  const int b    = blockIdx.z;
  const int tid  = threadIdx.x;
  const int w    = tid >> 5;
  const int lane = tid & 31;
  const int ln   = lane & 15;
  const int rb   = (lane >> 4) << 3;

  __shared__ alignas(16) h16 Ks[64 * 64];       // K tile,   [tk][d]
  __shared__ alignas(16) h16 Vt[64 * 64];       // V tile^T, [d][tk]
  __shared__ alignas(16) h16 Ps[4][16 * 64];    // per-wave P, [qrow][tk]
  __shared__ float msk[64];

  const size_t rs = 3 * N_EMBD;  // qkv row stride
  const int q0 = qt * 64 + w * 16;

  // Q fragments (16 rows x 64 d = 2 frags), held in registers for all K tiles
  const h16* qb = qkv + ((size_t)b * SEQ + q0) * rs + (size_t)h * HEAD_D;
  v16h qf[2];
  qf[0] = load_frag(qb, (int)rs);
  qf[1] = load_frag(qb + 32, (int)rs);

  v8f o[4];
#pragma unroll
  for (int d = 0; d < 4; ++d) o[d] = (v8f){0, 0, 0, 0, 0, 0, 0, 0};
  float mrow[8], lrow[8];
#pragma unroll
  for (int r = 0; r < 8; ++r) { mrow[r] = -1e30f; lrow[r] = 0.f; }

  for (int kt = 0; kt <= qt; ++kt) {  // causal: skip tiles fully above diagonal
    const int kt0 = kt * 64;
    // ---- stage K (async to LDS, row-major) and V (transposed) tiles ----
#pragma unroll
    for (int c = 0; c < 4; ++c) {
      int idx  = tid + c * 128;  // 0..511
      int row  = idx >> 3;
      int doff = (idx & 7) * 8;
      const h16* kp = qkv + ((size_t)b * SEQ + kt0 + row) * rs + N_EMBD +
                      (size_t)h * HEAD_D + doff;
      stage16(Ks + row * 64 + doff, kp);
      v8h vv = *(const v8h*)(kp + N_EMBD);
#pragma unroll
      for (int i = 0; i < 8; ++i) Vt[(doff + i) * 64 + row] = vv[i];
    }
    if (tid < 64)
      msk[tid] = (amask[(size_t)b * SEQ + kt0 + tid] != 0) ? 0.f : -1e30f;
    wait_async_lds();
    __syncthreads();

    // ---- S = (Q K^T) * scale, masked; track row max ----
    float p[4][8];
    float rmax[8];
#pragma unroll
    for (int r = 0; r < 8; ++r) rmax[r] = -1e30f;
#pragma unroll
    for (int nf = 0; nf < 4; ++nf) {
      v16h kf0 = load_frag(Ks + nf * 16 * 64, 64);
      v16h kf1 = load_frag(Ks + nf * 16 * 64 + 32, 64);
      v8f s = (v8f){0, 0, 0, 0, 0, 0, 0, 0};
      s = __builtin_amdgcn_wmma_f32_16x16x32_f16(false, qf[0], false, kf0,
                                                 (short)0, s, false, false);
      s = __builtin_amdgcn_wmma_f32_16x16x32_f16(false, qf[1], false, kf1,
                                                 (short)0, s, false, false);
      const int kcol = kt0 + nf * 16 + ln;
      const float madd = msk[nf * 16 + ln];
#pragma unroll
      for (int r = 0; r < 8; ++r) {
        int qrow = q0 + rb + r;
        float v = s[r] * 0.125f + madd;  // scale = 1/sqrt(64)
        if (kcol > qrow) v = -1e30f;     // causal
        p[nf][r] = v;
        rmax[r] = fmaxf(rmax[r], v);
      }
    }
    // reduce row max across the 16 lanes holding one row
    for (int off = 1; off < 16; off <<= 1)
#pragma unroll
      for (int r = 0; r < 8; ++r)
        rmax[r] = fmaxf(rmax[r], __shfl_xor(rmax[r], off, 32));

    // ---- online softmax update ----
    float corr[8], rsum[8];
#pragma unroll
    for (int r = 0; r < 8; ++r) {
      float mn = fmaxf(mrow[r], rmax[r]);
      corr[r] = __expf(mrow[r] - mn);
      mrow[r] = mn;
      rsum[r] = 0.f;
    }
#pragma unroll
    for (int nf = 0; nf < 4; ++nf)
#pragma unroll
      for (int r = 0; r < 8; ++r) {
        float e = __expf(p[nf][r] - mrow[r]);
        p[nf][r] = e;
        rsum[r] += e;
      }
    for (int off = 1; off < 16; off <<= 1)
#pragma unroll
      for (int r = 0; r < 8; ++r) rsum[r] += __shfl_xor(rsum[r], off, 32);
#pragma unroll
    for (int r = 0; r < 8; ++r) lrow[r] = lrow[r] * corr[r] + rsum[r];
#pragma unroll
    for (int d = 0; d < 4; ++d)
#pragma unroll
      for (int r = 0; r < 8; ++r) o[d][r] *= corr[r];

    // ---- P: C-layout regs -> row-major LDS (f16) for use as WMMA A ----
#pragma unroll
    for (int nf = 0; nf < 4; ++nf)
#pragma unroll
      for (int r = 0; r < 8; ++r)
        Ps[w][(rb + r) * 64 + nf * 16 + ln] = (h16)p[nf][r];
    __syncthreads();

    // ---- O += P V ----
#pragma unroll
    for (int d = 0; d < 4; ++d)
#pragma unroll
      for (int ks = 0; ks < 2; ++ks) {
        v16h pf = load_frag(Ps[w] + ks * 32, 64);
        v16h vf = load_frag(Vt + d * 16 * 64 + ks * 32, 64);
        o[d] = __builtin_amdgcn_wmma_f32_16x16x32_f16(false, pf, false, vf,
                                                      (short)0, o[d], false,
                                                      false);
      }
    __syncthreads();  // protect Ks/Vt/Ps before next tile restage
  }

  // ---- epilogue: O / l -> y (f16, [B][T][C] with c = h*64+d) ----
#pragma unroll
  for (int r = 0; r < 8; ++r) {
    float inv = 1.f / lrow[r];
    int qrow = q0 + rb + r;
    h16* yp = y + ((size_t)b * SEQ + qrow) * N_EMBD + (size_t)h * HEAD_D;
#pragma unroll
    for (int d = 0; d < 4; ++d) yp[d * 16 + ln] = (h16)(o[d][r] * inv);
  }
}

// ---------------------------------------------------------------------------
extern "C" void kernel_launch(void* const* d_in, const int* in_sizes, int n_in,
                              void* d_out, int out_size, void* d_ws,
                              size_t ws_size, hipStream_t stream) {
  const float* x      = (const float*)d_in[0];
  const int*   amask  = (const int*)d_in[1];
  const float* W_attn = (const float*)d_in[2];
  const float* W_proj = (const float*)d_in[3];
  float* out = (float*)d_out;

  const size_t M = (size_t)BATCH * SEQ;  // 8192 rows
  h16* Xh  = (h16*)d_ws;                         //  8M h16 = 16 MB
  h16* Wah = Xh  + M * N_EMBD;                   //  3M h16 =  6 MB
  h16* Wph = Wah + (size_t)3 * N_EMBD * N_EMBD;  //  1M h16 =  2 MB
  h16* QKV = Wph + (size_t)N_EMBD * N_EMBD;      // 24M h16 = 48 MB
  h16* Yh  = QKV + M * 3 * N_EMBD;               //  8M h16 = 16 MB  (88 MB tot)

  cvt_f32_f16<<<1024, 256, 0, stream>>>(x, Xh, M * N_EMBD);
  transpose_cvt<<<dim3(3 * N_EMBD / 32, N_EMBD / 32), dim3(32, 8), 0, stream>>>(
      W_attn, Wah, N_EMBD, 3 * N_EMBD);
  transpose_cvt<<<dim3(N_EMBD / 32, N_EMBD / 32), dim3(32, 8), 0, stream>>>(
      W_proj, Wph, N_EMBD, N_EMBD);

  gemm_f16<h16><<<dim3(3 * N_EMBD / 128, M / 128), 256, 0, stream>>>(
      Xh, Wah, QKV, (int)M, 3 * N_EMBD, N_EMBD);

  flash_attn<<<dim3(SEQ / 64, N_HEAD, BATCH), 128, 0, stream>>>(QKV, amask, Yh);

  gemm_f16<float><<<dim3(N_EMBD / 128, M / 128), 256, 0, stream>>>(
      Yh, Wph, out, (int)M, N_EMBD, N_EMBD);
}
